// AnyNet_8813272891699
// MI455X (gfx1250) — compile-verified, matrix operand order checked by reading, not run on hardware
//
#include <hip/hip_runtime.h>
#include <hip/hip_bf16.h>

// ---------------------------------------------------------------------------
// Types for CDNA5 WMMA (wave32): v_wmma_f32_16x16x32_f16
// ---------------------------------------------------------------------------
typedef __attribute__((ext_vector_type(8)))  _Float16 half8;
typedef __attribute__((ext_vector_type(16))) _Float16 v16h;
typedef __attribute__((ext_vector_type(8)))  float    v8f;

// ---------------------------------------------------------------------------
// Problem constants (from setup_inputs)
// ---------------------------------------------------------------------------
#define NB 8
#define H0 512
#define W0 960
#define H2 256
#define W2 480
#define H4 128
#define W4 240
#define H8 64
#define W8 120
#define H16 32
#define W16 60

static inline int ceil_div(int a, int b) { return (a + b - 1) / b; }

// ---------------------------------------------------------------------------
// 2D conv, k=3, stride=2, pad=1, + ReLU (feature pyramid, tiny channels)
// in:[B,CIN,Hin,Win] w:[COUT,CIN,3,3] out:[B,COUT,Hout,Wout]
// ---------------------------------------------------------------------------
template <int CIN, int COUT>
__global__ __launch_bounds__(256) void conv2d_s2_relu(
    const float* __restrict__ in, const float* __restrict__ w,
    float* __restrict__ out, int Hin, int Win, int Hout, int Wout) {
  int idx = blockIdx.x * blockDim.x + threadIdx.x;
  int total = NB * Hout * Wout;
  if (idx >= total) return;
  int ox = idx % Wout;
  int t = idx / Wout;
  int oy = t % Hout;
  int b = t / Hout;

  float acc[COUT];
#pragma unroll
  for (int c = 0; c < COUT; ++c) acc[c] = 0.f;

#pragma unroll
  for (int ci = 0; ci < CIN; ++ci) {
    const float* ip = in + ((long)(b * CIN + ci) * Hin) * Win;
#pragma unroll
    for (int ky = 0; ky < 3; ++ky) {
      int iy = 2 * oy + ky - 1;
      if (iy < 0 || iy >= Hin) continue;
#pragma unroll
      for (int kx = 0; kx < 3; ++kx) {
        int ix = 2 * ox + kx - 1;
        if (ix < 0 || ix >= Win) continue;
        float v = ip[(long)iy * Win + ix];
#pragma unroll
        for (int co = 0; co < COUT; ++co)
          acc[co] += w[((co * CIN + ci) * 3 + ky) * 3 + kx] * v;
      }
    }
  }
#pragma unroll
  for (int co = 0; co < COUT; ++co)
    out[((long)(b * COUT + co) * Hout + oy) * Wout + ox] = fmaxf(acc[co], 0.f);
}

// ---------------------------------------------------------------------------
// Full cost volume at 1/16: cost[b,d,h,w] = sum_c |ref - tgt_shift|
// ref,tgt:[B,8,H,W]  cost:[B,12,H,W]
// ---------------------------------------------------------------------------
__global__ __launch_bounds__(256) void build_vol12(
    const float* __restrict__ ref, const float* __restrict__ tgt,
    float* __restrict__ cost, int Hc, int Wc) {
  int idx = blockIdx.x * blockDim.x + threadIdx.x;
  int total = NB * 12 * Hc * Wc;
  if (idx >= total) return;
  int x = idx % Wc;
  int t = idx / Wc;
  int y = t % Hc;
  t /= Hc;
  int d = t % 12;
  int b = t / 12;
  int xs = x - d;
  float s = 0.f;
#pragma unroll
  for (int c = 0; c < 8; ++c) {
    long base = ((long)(b * 8 + c) * Hc + y) * Wc;
    float r = ref[base + x];
    float tv = (xs >= 0) ? tgt[base + xs] : 0.f;
    s += fabsf(r - tv);
  }
  cost[idx] = s;
}

// ---------------------------------------------------------------------------
// Residual cost volume (warp fused): offsets d-2, d=0..4
// ref,tgt:[B,8,H,W] disp:[B,1,H,W] cost:[B,5,H,W]
// ---------------------------------------------------------------------------
__global__ __launch_bounds__(256) void residual_vol(
    const float* __restrict__ ref, const float* __restrict__ tgt,
    const float* __restrict__ disp, float* __restrict__ cost, int Hc, int Wc) {
  int idx = blockIdx.x * blockDim.x + threadIdx.x;
  int total = NB * 5 * Hc * Wc;
  if (idx >= total) return;
  int x = idx % Wc;
  int t = idx / Wc;
  int y = t % Hc;
  t /= Hc;
  int d = t % 5;
  int b = t / 5;

  float o = (float)(d - 2);
  float dsp = disp[((long)b * Hc + y) * Wc + x] + o;
  float xf = (float)x - dsp;
  float x0f = floorf(xf);
  int x0 = (int)x0f;
  int x1 = x0 + 1;
  float w1 = xf - x0f;
  float w0 = 1.f - w1;
  float m0 = (x0 >= 0 && x0 <= Wc - 1) ? 1.f : 0.f;
  float m1 = (x1 >= 0 && x1 <= Wc - 1) ? 1.f : 0.f;
  int i0 = min(max(x0, 0), Wc - 1);
  int i1 = min(max(x1, 0), Wc - 1);
  float s = 0.f;
#pragma unroll
  for (int c = 0; c < 8; ++c) {
    long base = ((long)(b * 8 + c) * Hc + y) * Wc;
    float r = ref[base + x];
    float rec = tgt[base + i0] * (w0 * m0) + tgt[base + i1] * (w1 * m1);
    s += fabsf(r - rec);
  }
  cost[idx] = s;
}

// ---------------------------------------------------------------------------
// Generic direct 3D conv k=3 pad=1 (4-channel blocks at 1/8 and 1/4)
// in:[B,CIN,D,H,W] w:[COUT,CIN,3,3,3] out:[B,COUT,D,H,W]
// ---------------------------------------------------------------------------
template <int CIN, int COUT, bool RELU>
__global__ __launch_bounds__(256) void conv3d_direct(
    const float* __restrict__ in, const float* __restrict__ w,
    float* __restrict__ out, int Dd, int Hc, int Wc) {
  int idx = blockIdx.x * blockDim.x + threadIdx.x;
  int total = NB * COUT * Dd * Hc * Wc;
  if (idx >= total) return;
  int x = idx % Wc;
  int t = idx / Wc;
  int y = t % Hc;
  t /= Hc;
  int z = t % Dd;
  t /= Dd;
  int co = t % COUT;
  int b = t / COUT;

  float acc = 0.f;
#pragma unroll
  for (int ci = 0; ci < CIN; ++ci) {
    const float* ip = in + (long)(b * CIN + ci) * Dd * Hc * Wc;
    const float* wp = w + (co * CIN + ci) * 27;
#pragma unroll
    for (int dz = 0; dz < 3; ++dz) {
      int zz = z + dz - 1;
      if (zz < 0 || zz >= Dd) continue;
#pragma unroll
      for (int dy = 0; dy < 3; ++dy) {
        int yy = y + dy - 1;
        if (yy < 0 || yy >= Hc) continue;
#pragma unroll
        for (int dx = 0; dx < 3; ++dx) {
          int xx = x + dx - 1;
          if (xx < 0 || xx >= Wc) continue;
          acc += wp[dz * 9 + dy * 3 + dx] * ip[((long)zz * Hc + yy) * Wc + xx];
        }
      }
    }
  }
  if (RELU) acc = fmaxf(acc, 0.f);
  out[idx] = acc;
}

// ---------------------------------------------------------------------------
// Block-1 layer 0: 1 -> 16 channels + ReLU, output f16 channels-last
// in:[B,1,D,H,W] f32   w:[16,1,27]   out:[b][z][y][x][c16] f16
// ---------------------------------------------------------------------------
__global__ __launch_bounds__(256) void conv3d_1to16_cl(
    const float* __restrict__ in, const float* __restrict__ w,
    _Float16* __restrict__ out, int Dd, int Hc, int Wc) {
  int idx = blockIdx.x * blockDim.x + threadIdx.x;
  int total = NB * Dd * Hc * Wc;
  if (idx >= total) return;
  int x = idx % Wc;
  int t = idx / Wc;
  int y = t % Hc;
  t /= Hc;
  int z = t % Dd;
  int b = t / Dd;

  float acc[16];
#pragma unroll
  for (int c = 0; c < 16; ++c) acc[c] = 0.f;
#pragma unroll
  for (int dz = 0; dz < 3; ++dz) {
    int zz = z + dz - 1;
    if (zz < 0 || zz >= Dd) continue;
#pragma unroll
    for (int dy = 0; dy < 3; ++dy) {
      int yy = y + dy - 1;
      if (yy < 0 || yy >= Hc) continue;
#pragma unroll
      for (int dx = 0; dx < 3; ++dx) {
        int xx = x + dx - 1;
        if (xx < 0 || xx >= Wc) continue;
        float v = in[((long)(b * Dd + zz) * Hc + yy) * Wc + xx];
        int tap = dz * 9 + dy * 3 + dx;
#pragma unroll
        for (int co = 0; co < 16; ++co) acc[co] += w[co * 27 + tap] * v;
      }
    }
  }
  _Float16* op = out + (long)idx * 16;
#pragma unroll
  for (int co = 0; co < 16; ++co) op[co] = (_Float16)fmaxf(acc[co], 0.f);
}

// ---------------------------------------------------------------------------
// Block-1 layer 5: 16 -> 1 channel, no ReLU, input f16 channels-last
// in:[b][z][y][x][c16] f16   w:[1,16,27]   out:[B,1,D,H,W] f32
// ---------------------------------------------------------------------------
__global__ __launch_bounds__(256) void conv3d_16to1_cl(
    const _Float16* __restrict__ in, const float* __restrict__ w,
    float* __restrict__ out, int Dd, int Hc, int Wc) {
  int idx = blockIdx.x * blockDim.x + threadIdx.x;
  int total = NB * Dd * Hc * Wc;
  if (idx >= total) return;
  int x = idx % Wc;
  int t = idx / Wc;
  int y = t % Hc;
  t /= Hc;
  int z = t % Dd;
  int b = t / Dd;

  float acc = 0.f;
#pragma unroll
  for (int dz = 0; dz < 3; ++dz) {
    int zz = z + dz - 1;
    if (zz < 0 || zz >= Dd) continue;
#pragma unroll
    for (int dy = 0; dy < 3; ++dy) {
      int yy = y + dy - 1;
      if (yy < 0 || yy >= Hc) continue;
#pragma unroll
      for (int dx = 0; dx < 3; ++dx) {
        int xx = x + dx - 1;
        if (xx < 0 || xx >= Wc) continue;
        int tap = dz * 9 + dy * 3 + dx;
        const _Float16* ip = in + ((long)(b * Dd + zz) * Hc + yy) * Wc * 16 + (long)xx * 16;
#pragma unroll
        for (int ci = 0; ci < 16; ++ci) acc += (float)ip[ci] * w[ci * 27 + tap];
      }
    }
  }
  out[idx] = acc;
}

// ---------------------------------------------------------------------------
// Pack 16x16x3x3x3 f32 weights into per-lane WMMA B-fragments (f16).
// bp[(s*32+lane)*16 + e]; K(e,lane) = s*32 + (e<8?0:16) + (lane/16)*8 + e%8
// B element is W[co = lane%16][ci = K%16][tap = K/16]; K>=432 -> 0.
// ---------------------------------------------------------------------------
__global__ __launch_bounds__(256) void pack_w16(const float* __restrict__ w,
                                                _Float16* __restrict__ bp) {
  int i = blockIdx.x * blockDim.x + threadIdx.x;
  if (i >= 14 * 32 * 16) return;
  int e = i % 16;
  int L = (i / 16) % 32;
  int s = i / (16 * 32);
  int K = s * 32 + ((e < 8) ? 0 : 16) + (L / 16) * 8 + (e % 8);
  int co = L % 16;
  float v = 0.f;
  if (K < 432) {
    int tap = K / 16;
    int ci = K % 16;
    v = w[(co * 16 + ci) * 27 + tap];
  }
  bp[i] = (_Float16)v;
}

// ---------------------------------------------------------------------------
// WMMA implicit-GEMM 3D conv: 16 -> 16 channels, k=3, pad=1, + ReLU.
// f16 channels-last in/out. Workgroup = 8 waves sharing one LDS halo
// (3z x 10y x 18x x 16c); wave w computes y-row ybase+w as a 16x16 tile.
// Halo fetched with GLOBAL_LOAD_ASYNC_TO_LDS_B128 (ASYNCcnt), OOB chunks
// zero-filled with ds_store_b128 (disjoint addresses). 14 x v_wmma per wave.
// ---------------------------------------------------------------------------
__global__ __launch_bounds__(256) void conv3d16_wmma(
    const _Float16* __restrict__ in, const _Float16* __restrict__ bp,
    _Float16* __restrict__ out, int Dd, int Hc, int Wc) {
  __shared__ __align__(16) _Float16 tile[3 * 10 * 18 * 16];  // 8640 f16 = 17.3KB

  int tid = threadIdx.x;
  int lane = tid & 31;
  int wv = tid >> 5;  // wave id 0..7 -> y row within block
  int ntx = (Wc + 15) / 16;
  int nty = Hc / 8;
  int t = blockIdx.x;
  int tx = t % ntx;
  t /= ntx;
  int ty = t % nty;
  t /= nty;
  int z0 = t % Dd;
  int b = t / Dd;
  int x0 = tx * 16;
  int ybase = ty * 8;

  // ---- async halo load: 1080 chunks of 8 f16 (16B each) ----
  unsigned ldsbase = (unsigned)(size_t)(&tile[0]);
  half8 zero8 = {};
  for (int i = tid; i < 1080; i += 256) {
    int c8 = i & 1;
    int r = i >> 1;
    int xx = r % 18;
    r /= 18;
    int yy = r % 10;
    int zz = r / 10;
    int z = z0 + zz - 1, y = ybase + yy - 1, x = x0 + xx - 1;
    int lidx = ((zz * 10 + yy) * 18 + xx) * 16 + c8 * 8;
    if (z >= 0 && z < Dd && y >= 0 && y < Hc && x >= 0 && x < Wc) {
      unsigned loff = ldsbase + (unsigned)(lidx * 2);
      const _Float16* gp =
          in + (((long)(b * Dd + z) * Hc + y) * Wc + x) * 16 + c8 * 8;
      asm volatile("global_load_async_to_lds_b128 %0, %1, off" ::"v"(loff),
                   "v"((unsigned long long)(size_t)gp)
                   : "memory");
    } else {
      *(half8*)&tile[lidx] = zero8;  // zero-pad boundary chunk
    }
  }
  asm volatile("s_wait_asynccnt 0x0" ::: "memory");
  __syncthreads();

  // ---- 14 K-steps of v_wmma_f32_16x16x32_f16 ----
  int M = lane & 15;  // A row (x in tile) == D column (co)
  int g = lane >> 4;  // K group
  int c0 = g * 8;     // ci base of this lane's K runs
  v8f acc = {};
  for (int s = 0; s < 14; ++s) {
    int tap0 = 2 * s;      // wave-uniform
    int tap1 = 2 * s + 1;  // wave-uniform; 27 => zero pad
    union {
      v16h v;
      half8 h[2];
    } A;
    {
      int dz = tap0 / 9, rr = tap0 % 9, dy = rr / 3, dx = rr % 3;
      A.h[0] = *(const half8*)&tile[((dz * 10 + wv + dy) * 18 + (M + dx)) * 16 + c0];
    }
    if (tap1 < 27) {
      int dz = tap1 / 9, rr = tap1 % 9, dy = rr / 3, dx = rr % 3;
      A.h[1] = *(const half8*)&tile[((dz * 10 + wv + dy) * 18 + (M + dx)) * 16 + c0];
    } else {
      A.h[1] = zero8;
    }
    v16h Bf = *(const v16h*)(bp + (s * 32 + lane) * 16);
    acc = __builtin_amdgcn_wmma_f32_16x16x32_f16(false, A.v, false, Bf,
                                                 (short)0, acc, false, false);
  }

  // ---- D: lane%16 = co; VGPR r -> x offset r + 8*(lane/16). ReLU, f16 CL ----
  int y = ybase + wv;
  int co = M;
  _Float16* ob = out + (((long)(b * Dd + z0) * Hc + y) * Wc) * 16 + co;
#pragma unroll
  for (int r = 0; r < 8; ++r) {
    int xm = x0 + r + g * 8;
    if (xm < Wc) ob[(long)xm * 16] = (_Float16)fmaxf(acc[r], 0.f);
  }
}

// ---------------------------------------------------------------------------
// Fused softmax(-cost) over D + disparity regression (* up scale).
// cost:[B,D,H,W] -> disp:[B,1,H,W];  disp = scale * sum_d p_d*(start+d)
// ---------------------------------------------------------------------------
__global__ __launch_bounds__(256) void softmax_dispreg(
    const float* __restrict__ cost, float* __restrict__ disp, int Dd, int Hc,
    int Wc, float start, float scale) {
  int idx = blockIdx.x * blockDim.x + threadIdx.x;
  int total = NB * Hc * Wc;
  if (idx >= total) return;
  int x = idx % Wc;
  int t = idx / Wc;
  int y = t % Hc;
  int b = t / Hc;
  long base = ((long)b * Dd * Hc + y) * Wc + x;
  long stride = (long)Hc * Wc;
  float mx = -3.4e38f;
  for (int d = 0; d < Dd; ++d) mx = fmaxf(mx, -cost[base + d * stride]);
  float se = 0.f, sw = 0.f;
  for (int d = 0; d < Dd; ++d) {
    float e = expf(-cost[base + d * stride] - mx);
    se += e;
    sw += e * (start + (float)d);
  }
  disp[idx] = scale * sw / se;
}

// ---------------------------------------------------------------------------
// Bilinear resize (align_corners=True), C=1, fused *mul and optional +add.
// ---------------------------------------------------------------------------
__global__ __launch_bounds__(256) void resize_bl(
    const float* __restrict__ in, float* __restrict__ out, int Hin, int Win,
    int Hout, int Wout, float mul, const float* __restrict__ addsrc) {
  int idx = blockIdx.x * blockDim.x + threadIdx.x;
  int total = NB * Hout * Wout;
  if (idx >= total) return;
  int ox = idx % Wout;
  int t = idx / Wout;
  int oy = t % Hout;
  int b = t / Hout;
  float sy = (Hout > 1) ? (float)(Hin - 1) / (float)(Hout - 1) : 0.f;
  float sx = (Wout > 1) ? (float)(Win - 1) / (float)(Wout - 1) : 0.f;
  float yf = oy * sy;
  int y0 = (int)floorf(yf);
  int y1 = min(y0 + 1, Hin - 1);
  float wy = yf - (float)y0;
  float xf = ox * sx;
  int x0 = (int)floorf(xf);
  int x1 = min(x0 + 1, Win - 1);
  float wx = xf - (float)x0;
  const float* ib = in + (long)b * Hin * Win;
  float r0 = ib[(long)y0 * Win + x0] * (1.f - wy) + ib[(long)y1 * Win + x0] * wy;
  float r1 = ib[(long)y0 * Win + x1] * (1.f - wy) + ib[(long)y1 * Win + x1] * wy;
  float v = (r0 * (1.f - wx) + r1 * wx) * mul;
  if (addsrc) v += addsrc[idx];
  out[idx] = v;
}

// ---------------------------------------------------------------------------
// Host orchestration
// ---------------------------------------------------------------------------
extern "C" void kernel_launch(void* const* d_in, const int* in_sizes, int n_in,
                              void* d_out, int out_size, void* d_ws,
                              size_t ws_size, hipStream_t stream) {
  (void)in_sizes; (void)n_in; (void)out_size; (void)ws_size;
  const float* left  = (const float*)d_in[0];
  const float* right = (const float*)d_in[1];
  const float* fw1 = (const float*)d_in[2];
  const float* fw2 = (const float*)d_in[3];
  const float* fw3 = (const float*)d_in[4];
  const float* fw4 = (const float*)d_in[5];
  const float* c1[6];  for (int j = 0; j < 6; ++j) c1[j] = (const float*)d_in[6 + j];
  const float* c2[6];  for (int j = 0; j < 6; ++j) c2[j] = (const float*)d_in[12 + j];
  const float* c3[6];  for (int j = 0; j < 6; ++j) c3[j] = (const float*)d_in[18 + j];

  char* wsp = (char*)d_ws;
  auto alloc = [&](size_t nbytes) -> void* {
    void* p = (void*)wsp;
    wsp += (nbytes + 255) & ~(size_t)255;
    return p;
  };

  float* xl   = (float*)alloc((size_t)NB * 4 * H2 * W2 * 4);
  float* xr   = (float*)alloc((size_t)NB * 4 * H2 * W2 * 4);
  float* f4l  = (float*)alloc((size_t)NB * 8 * H4 * W4 * 4);
  float* f4r  = (float*)alloc((size_t)NB * 8 * H4 * W4 * 4);
  float* f8l  = (float*)alloc((size_t)NB * 8 * H8 * W8 * 4);
  float* f8r  = (float*)alloc((size_t)NB * 8 * H8 * W8 * 4);
  float* f16l = (float*)alloc((size_t)NB * 8 * H16 * W16 * 4);
  float* f16r = (float*)alloc((size_t)NB * 8 * H16 * W16 * 4);
  float* cost0 = (float*)alloc((size_t)NB * 12 * H16 * W16 * 4);
  _Float16* volA = (_Float16*)alloc((size_t)NB * 16 * 12 * H16 * W16 * 2);
  _Float16* volB = (_Float16*)alloc((size_t)NB * 16 * 12 * H16 * W16 * 2);
  _Float16* bpk[4];
  for (int j = 0; j < 4; ++j) bpk[j] = (_Float16*)alloc((size_t)14 * 32 * 16 * 2);
  float* disp0 = (float*)alloc((size_t)NB * H16 * W16 * 4);
  float* warp1 = (float*)alloc((size_t)NB * H8 * W8 * 4);
  float* cost1 = (float*)alloc((size_t)NB * 5 * H8 * W8 * 4);
  float* v1a   = (float*)alloc((size_t)NB * 4 * 5 * H8 * W8 * 4);
  float* v1b   = (float*)alloc((size_t)NB * 4 * 5 * H8 * W8 * 4);
  float* disp1 = (float*)alloc((size_t)NB * H8 * W8 * 4);
  float* warp2 = (float*)alloc((size_t)NB * H4 * W4 * 4);
  float* cost2 = (float*)alloc((size_t)NB * 5 * H4 * W4 * 4);
  float* v2a   = (float*)alloc((size_t)NB * 4 * 5 * H4 * W4 * 4);
  float* v2b   = (float*)alloc((size_t)NB * 4 * 5 * H4 * W4 * 4);
  float* disp2 = (float*)alloc((size_t)NB * H4 * W4 * 4);

  const long NP = (long)NB * H0 * W0;
  float* pred0 = (float*)d_out;
  float* pred1 = (float*)d_out + NP;
  float* pred2 = (float*)d_out + 2 * NP;

  const int TB = 256;
#define G1(n) dim3(ceil_div((n), TB)), dim3(TB), 0, stream

  // ---- feature pyramid (both images) ----
  conv2d_s2_relu<3, 4><<<G1(NB * H2 * W2)>>>(left, fw1, xl, H0, W0, H2, W2);
  conv2d_s2_relu<3, 4><<<G1(NB * H2 * W2)>>>(right, fw1, xr, H0, W0, H2, W2);
  conv2d_s2_relu<4, 8><<<G1(NB * H4 * W4)>>>(xl, fw2, f4l, H2, W2, H4, W4);
  conv2d_s2_relu<4, 8><<<G1(NB * H4 * W4)>>>(xr, fw2, f4r, H2, W2, H4, W4);
  conv2d_s2_relu<8, 8><<<G1(NB * H8 * W8)>>>(f4l, fw3, f8l, H4, W4, H8, W8);
  conv2d_s2_relu<8, 8><<<G1(NB * H8 * W8)>>>(f4r, fw3, f8r, H4, W4, H8, W8);
  conv2d_s2_relu<8, 8><<<G1(NB * H16 * W16)>>>(f8l, fw4, f16l, H8, W8, H16, W16);
  conv2d_s2_relu<8, 8><<<G1(NB * H16 * W16)>>>(f8r, fw4, f16r, H8, W8, H16, W16);

  // ---- stage 0: full cost volume + WMMA 3D conv block (hidden 16) ----
  build_vol12<<<G1(NB * 12 * H16 * W16)>>>(f16l, f16r, cost0, H16, W16);
  for (int j = 0; j < 4; ++j)
    pack_w16<<<G1(14 * 32 * 16)>>>(c1[1 + j], bpk[j]);

  conv3d_1to16_cl<<<G1(NB * 12 * H16 * W16)>>>(cost0, c1[0], volA, 12, H16, W16);
  {
    int blocks = NB * 12 * (H16 / 8) * ceil_div(W16, 16);
    conv3d16_wmma<<<dim3(blocks), dim3(256), 0, stream>>>(volA, bpk[0], volB, 12, H16, W16);
    conv3d16_wmma<<<dim3(blocks), dim3(256), 0, stream>>>(volB, bpk[1], volA, 12, H16, W16);
    conv3d16_wmma<<<dim3(blocks), dim3(256), 0, stream>>>(volA, bpk[2], volB, 12, H16, W16);
    conv3d16_wmma<<<dim3(blocks), dim3(256), 0, stream>>>(volB, bpk[3], volA, 12, H16, W16);
  }
  conv3d_16to1_cl<<<G1(NB * 12 * H16 * W16)>>>(volA, c1[5], cost0, 12, H16, W16);
  softmax_dispreg<<<G1(NB * H16 * W16)>>>(cost0, disp0, 12, H16, W16, 0.f,
                                          (float)H0 / (float)H16);
  resize_bl<<<G1(NB * H0 * W0)>>>(disp0, pred0, H16, W16, H0, W0, 1.f, nullptr);

  // ---- stage 1: residual volume at 1/8 (hidden 4, direct conv) ----
  resize_bl<<<G1(NB * H8 * W8)>>>(pred0, warp1, H0, W0, H8, W8,
                                  (float)H8 / (float)H0, nullptr);
  residual_vol<<<G1(NB * 5 * H8 * W8)>>>(f8l, f8r, warp1, cost1, H8, W8);
  conv3d_direct<1, 4, true><<<G1(NB * 4 * 5 * H8 * W8)>>>(cost1, c2[0], v1a, 5, H8, W8);
  conv3d_direct<4, 4, true><<<G1(NB * 4 * 5 * H8 * W8)>>>(v1a, c2[1], v1b, 5, H8, W8);
  conv3d_direct<4, 4, true><<<G1(NB * 4 * 5 * H8 * W8)>>>(v1b, c2[2], v1a, 5, H8, W8);
  conv3d_direct<4, 4, true><<<G1(NB * 4 * 5 * H8 * W8)>>>(v1a, c2[3], v1b, 5, H8, W8);
  conv3d_direct<4, 4, true><<<G1(NB * 4 * 5 * H8 * W8)>>>(v1b, c2[4], v1a, 5, H8, W8);
  conv3d_direct<4, 1, false><<<G1(NB * 5 * H8 * W8)>>>(v1a, c2[5], cost1, 5, H8, W8);
  softmax_dispreg<<<G1(NB * H8 * W8)>>>(cost1, disp1, 5, H8, W8, -2.f,
                                        (float)H0 / (float)H8);
  resize_bl<<<G1(NB * H0 * W0)>>>(disp1, pred1, H8, W8, H0, W0, 1.f, pred0);

  // ---- stage 2: residual volume at 1/4 (hidden 4, direct conv) ----
  resize_bl<<<G1(NB * H4 * W4)>>>(pred1, warp2, H0, W0, H4, W4,
                                  (float)H4 / (float)H0, nullptr);
  residual_vol<<<G1(NB * 5 * H4 * W4)>>>(f4l, f4r, warp2, cost2, H4, W4);
  conv3d_direct<1, 4, true><<<G1(NB * 4 * 5 * H4 * W4)>>>(cost2, c3[0], v2a, 5, H4, W4);
  conv3d_direct<4, 4, true><<<G1(NB * 4 * 5 * H4 * W4)>>>(v2a, c3[1], v2b, 5, H4, W4);
  conv3d_direct<4, 4, true><<<G1(NB * 4 * 5 * H4 * W4)>>>(v2b, c3[2], v2a, 5, H4, W4);
  conv3d_direct<4, 4, true><<<G1(NB * 4 * 5 * H4 * W4)>>>(v2a, c3[3], v2b, 5, H4, W4);
  conv3d_direct<4, 4, true><<<G1(NB * 4 * 5 * H4 * W4)>>>(v2b, c3[4], v2a, 5, H4, W4);
  conv3d_direct<4, 1, false><<<G1(NB * 5 * H4 * W4)>>>(v2a, c3[5], cost2, 5, H4, W4);
  softmax_dispreg<<<G1(NB * H4 * W4)>>>(cost2, disp2, 5, H4, W4, -2.f,
                                        (float)H0 / (float)H4);
  resize_bl<<<G1(NB * H0 * W0)>>>(disp2, pred2, H4, W4, H0, W0, 1.f, pred1);
#undef G1
}
